// TransducerAcousticModle_76390288327553
// MI455X (gfx1250) — compile-verified, matrix operand order checked by reading, not run on hardware
//
#include <hip/hip_runtime.h>
#include <hip/hip_bf16.h>

typedef __attribute__((ext_vector_type(16))) _Float16 v16h;
typedef __attribute__((ext_vector_type(8)))  _Float16 v8h;
typedef __attribute__((ext_vector_type(8)))  float    v8f;

#define B_     4
#define T_     400
#define U_     64
#define E_     320
#define INNER_ 512
#define VOCAB_ 1024

#define LDS_STRIDE (INNER_ + 8)   // halves; pad 8 halves (16B) to dodge bank conflicts

// tanh via hardware exp2/rcp (no ftanh libcall on gfx1250 toolchain).
// tanh(x) = (e^{2x}-1)/(e^{2x}+1); clamp so exp2 can't overflow -> no NaN.
__device__ __forceinline__ float fast_tanh(float x) {
  x = fminf(10.0f, fmaxf(-10.0f, x));
  const float LOG2E_X2 = 2.8853900817779268f;           // 2*log2(e)
  float e = __builtin_amdgcn_exp2f(x * LOG2E_X2);       // v_exp_f32
  return (e - 1.0f) * __builtin_amdgcn_rcpf(e + 1.0f);  // v_rcp_f32
}

// ---------------------------------------------------------------------------
// Kernel 1: P[r][i] = sum_k X[r][k] * W1[i][coloff + k]   (K = E = 320)
// Used for enc_proj (coloff=0) and dec_proj (coloff=E). Tiny GEMMs (<0.6 GFLOP
// combined) -> plain f32 FMA, vectorized float4 loads.
// ---------------------------------------------------------------------------
__global__ void proj_kernel(const float* __restrict__ X,
                            const float* __restrict__ W1,
                            float* __restrict__ P,
                            int rows, int coloff) {
  int idx = blockIdx.x * blockDim.x + threadIdx.x;
  if (idx >= rows * INNER_) return;
  int r = idx / INNER_;
  int i = idx - r * INNER_;
  const float4* x = (const float4*)(X + (size_t)r * E_);
  const float4* w = (const float4*)(W1 + (size_t)i * (2 * E_) + coloff);
  float acc = 0.0f;
#pragma unroll 4
  for (int k = 0; k < E_ / 4; ++k) {
    float4 a = x[k];
    float4 b = w[k];
    acc += a.x * b.x + a.y * b.y + a.z * b.z + a.w * b.w;
  }
  P[idx] = acc;
}

// ---------------------------------------------------------------------------
// Kernel 2: f32 -> f16 conversion for W2 (1024x512, 1 MB as f16; lives in L2)
// ---------------------------------------------------------------------------
__global__ void cvt_f16_kernel(const float* __restrict__ src,
                               _Float16* __restrict__ dst, int n) {
  int i = blockIdx.x * blockDim.x + threadIdx.x;
  if (i < n) dst[i] = (_Float16)src[i];
}

// ---------------------------------------------------------------------------
// Kernel 3 (main): per workgroup -> one (b,t) and a 16-wide u-block.
//  Stage 1: h[16][512] = tanh(encP[b,t,:] + decP[b,u,:] + b1) as f16 in LDS.
//  Stage 2: 8 waves; wave w covers VOCAB columns [w*128, w*128+128):
//           8 N-tiles x 16 K-tiles of v_wmma_f32_16x16x32_f16.
// ---------------------------------------------------------------------------
__global__ __launch_bounds__(256)
void joint_wmma_kernel(const float* __restrict__ encP,     // (B*T, INNER)
                       const float* __restrict__ decP,     // (B*U, INNER)
                       const float* __restrict__ bias1,    // (INNER)
                       const _Float16* __restrict__ W2h,   // (VOCAB, INNER)
                       const float* __restrict__ bias2,    // (VOCAB)
                       float* __restrict__ out) {          // (B,T,U,VOCAB)
  __shared__ __align__(16) _Float16 hk[16 * LDS_STRIDE];

  const int wg = blockIdx.x;        // ((b*T + t) * 4) + ublock
  const int ub = wg & 3;
  const int bt = wg >> 2;           // b*T + t
  const int b  = bt / T_;

  const int tid = threadIdx.x;

  // ---------------- Stage 1: build h tile (16 x 512) in LDS as f16 ---------
  {
    const int m  = tid >> 4;              // row within u-block: 0..15
    const int k0 = (tid & 15) * 32;       // 32 contiguous k per thread
    const float* encRow = encP + (size_t)bt * INNER_;
    const float* decRow = decP + ((size_t)b * U_ + (size_t)(ub * 16 + m)) * INNER_;
    _Float16* dst = hk + m * LDS_STRIDE + k0;
#pragma unroll
    for (int k = 0; k < 32; k += 4) {
      float4 e  = *(const float4*)(encRow + k0 + k);
      float4 d  = *(const float4*)(decRow + k0 + k);
      float4 bb = *(const float4*)(bias1  + k0 + k);
      dst[k + 0] = (_Float16)fast_tanh(e.x + d.x + bb.x);
      dst[k + 1] = (_Float16)fast_tanh(e.y + d.y + bb.y);
      dst[k + 2] = (_Float16)fast_tanh(e.z + d.z + bb.z);
      dst[k + 3] = (_Float16)fast_tanh(e.w + d.w + bb.w);
    }
  }
  __syncthreads();

  // ---------------- Stage 2: WMMA GEMM  (16 x 1024) = h(16x512) @ W2h^T ----
  const int wave = tid >> 5;
  const int lane = tid & 31;
  const int lm   = lane & 15;
  const int hiK  = (lane >> 4) * 8;   // 16-bit A layout: hi half-wave takes K+8

  // A fragment source: row lm of h, offset by hiK halves
  const _Float16* aBase = hk + lm * LDS_STRIDE + hiK;

  const size_t outBase = ((size_t)bt * U_ + (size_t)(ub * 16)) * VOCAB_;
  const int    mOff    = (lane >> 4) * 8;   // C/D layout: lanes 16-31 hold M+8

  for (int nt = 0; nt < 8; ++nt) {
    const int n0 = wave * 128 + nt * 16;
    // B operand: column N = n0+lm  ==  row (n0+lm) of W2h (row-major = K-contig)
    const _Float16* bRow = W2h + (size_t)(n0 + lm) * INNER_ + hiK;

    v8f acc = {};
#pragma unroll 4
    for (int kt = 0; kt < 16; ++kt) {
      const int kk = kt * 32;
      v8h a0 = *(const v8h*)(aBase + kk);        // K {0..7}   (+hiK)
      v8h a1 = *(const v8h*)(aBase + kk + 16);   // K {16..23} (+hiK)
      v8h b0 = *(const v8h*)(bRow  + kk);
      v8h b1 = *(const v8h*)(bRow  + kk + 16);
      v16h A  = __builtin_shufflevector(a0, a1, 0,1,2,3,4,5,6,7,8,9,10,11,12,13,14,15);
      v16h Bm = __builtin_shufflevector(b0, b1, 0,1,2,3,4,5,6,7,8,9,10,11,12,13,14,15);
      acc = __builtin_amdgcn_wmma_f32_16x16x32_f16(
          /*neg_a=*/false, A, /*neg_b=*/false, Bm,
          /*c_mod=*/(short)0, acc, /*reuse_a=*/false, /*reuse_b=*/false);
    }

    const float bv = bias2[n0 + lm];
    float* o = out + outBase + (size_t)mOff * VOCAB_ + (size_t)(n0 + lm);
#pragma unroll
    for (int r = 0; r < 8; ++r)
      o[(size_t)r * VOCAB_] = acc[r] + bv;      // M = mOff + r, N = n0 + lm
  }
}

// ---------------------------------------------------------------------------
extern "C" void kernel_launch(void* const* d_in, const int* in_sizes, int n_in,
                              void* d_out, int out_size, void* d_ws, size_t ws_size,
                              hipStream_t stream) {
  const float* enc = (const float*)d_in[0];   // (B,T,E)
  const float* dec = (const float*)d_in[1];   // (B,U,E)
  const float* W1  = (const float*)d_in[2];   // (INNER, 2E)
  const float* b1  = (const float*)d_in[3];   // (INNER)
  const float* W2  = (const float*)d_in[4];   // (VOCAB, INNER)
  const float* b2  = (const float*)d_in[5];   // (VOCAB)
  float* out = (float*)d_out;

  // Workspace layout (all 16B aligned):
  //   encP : B*T*INNER f32   (3,276,800 B)
  //   decP : B*U*INNER f32   (  524,288 B)
  //   W2h  : VOCAB*INNER f16 (1,048,576 B)
  char* ws = (char*)d_ws;
  float*    encP = (float*)ws;
  float*    decP = (float*)(ws + (size_t)B_ * T_ * INNER_ * sizeof(float));
  _Float16* W2h  = (_Float16*)(ws + (size_t)B_ * T_ * INNER_ * sizeof(float)
                                  + (size_t)B_ * U_ * INNER_ * sizeof(float));

  // enc_proj: (B*T, INNER), K = E, W1 columns [0, E)
  {
    int rows = B_ * T_;
    int n = rows * INNER_;
    proj_kernel<<<(n + 255) / 256, 256, 0, stream>>>(enc, W1, encP, rows, 0);
  }
  // dec_proj: (B*U, INNER), K = E, W1 columns [E, 2E)
  {
    int rows = B_ * U_;
    int n = rows * INNER_;
    proj_kernel<<<(n + 255) / 256, 256, 0, stream>>>(dec, W1, decP, rows, E_);
  }
  // W2 -> f16
  {
    int n = VOCAB_ * INNER_;
    cvt_f16_kernel<<<(n + 255) / 256, 256, 0, stream>>>(W2, W2h, n);
  }
  // Fused tanh + WMMA GEMM + bias
  {
    int grid = B_ * T_ * (U_ / 16);   // 6400 workgroups, no divergence
    joint_wmma_kernel<<<grid, 256, 0, stream>>>(encP, decP, b1, W2h, b2, out);
  }
}
